// GGCNN_HNN_43379169689778
// MI455X (gfx1250) — matrix operand order repaired and checked by reading, not compiled, not run on hardware
//
#include <hip/hip_runtime.h>
#include <hip/hip_bf16.h>

// GConvGRU(K=1, H=None) stack == per-node MLP. edge_index, Wh1, Wh2 and the
// r-gate columns are dead (h=0). Core GEMM a1[500k,32] @ Wx2[32,{z,c blocks}]
// is routed through v_wmma_f32_16x16x32_f16 (K=32 == H1). The gate tail
// (3 transcendentals per hidden feature) uses gfx1250 v_tanh_f32 when the
// builtin is available.

typedef __attribute__((ext_vector_type(16))) _Float16 v16h;
typedef __attribute__((ext_vector_type(8)))  _Float16 v8h;
typedef __attribute__((ext_vector_type(2)))  _Float16 v2h;
typedef __attribute__((ext_vector_type(8)))  float    v8f;

#define H1v 32
#define H2v 128
#define WAVES_PER_BLOCK 4
#define BLOCK_THREADS 128

__device__ __forceinline__ float tanh_f(float x) {
#if __has_builtin(__builtin_amdgcn_tanhf)
    return __builtin_amdgcn_tanhf(x);           // v_tanh_f32 (gfx1250 TRANS op)
#elif __has_builtin(__builtin_amdgcn_tanh_f32)
    return __builtin_amdgcn_tanh_f32(x);
#else
    // tanh(x) = 1 - 2/(exp(2x)+1); fast rcp avoids precise-div expansion
    return fmaf(-2.0f, __builtin_amdgcn_rcpf(__expf(2.0f * x) + 1.0f), 1.0f);
#endif
}

// sigmoid(x) = 0.5*tanh(x/2) + 0.5 ; returns (1 - sigmoid(x)) directly since
// that's what the GRU update needs: h = (1-z)*c
__device__ __forceinline__ float one_minus_sigmoid_f(float x) {
#if __has_builtin(__builtin_amdgcn_tanhf) || __has_builtin(__builtin_amdgcn_tanh_f32)
    return fmaf(-0.5f, tanh_f(0.5f * x), 0.5f);
#else
    // 1 - 1/(1+e^-x) = 1/(1+e^x)
    return __builtin_amdgcn_rcpf(__expf(x) + 1.0f);
#endif
}

__global__ void __launch_bounds__(BLOCK_THREADS)
ggcnn_hnn_fused(const float* __restrict__ x,
                const float* __restrict__ Wx1, const float* __restrict__ bx1,
                const float* __restrict__ bh1,
                const float* __restrict__ Wx2, const float* __restrict__ bx2,
                const float* __restrict__ bh2,
                const float* __restrict__ lin_w, const float* __restrict__ lin_b,
                const float* __restrict__ grad_w, const float* __restrict__ grad_b,
                float* __restrict__ out,
                int nNodes, int nGroups) {
    // sW[sel][col][k]: sel 0 = z-gate cols (Wx2[:,0:128]), sel 1 = c-gate cols
    // (Wx2[:,256:384]); column-major over k so a lane's B-fragment is two 8-half
    // (16B) runs at k = 8*hi and 16 + 8*hi in LDS.
    __shared__ __align__(16) _Float16 sW[2 * H2v * H1v];             // 16 KB
    __shared__ __align__(16) _Float16 sA[WAVES_PER_BLOCK][16 * H1v]; // 4 KB
    __shared__ float sBz[H2v];      // bx2[c]+bh2[c]           (z gate)
    __shared__ float sBc[H2v];      // bx2[256+c]+bh2[256+c]   (c gate)
    __shared__ float sLw[H2v];      // lin_w
    __shared__ float sL1[6][H1v];   // layer-1: wz0,wz1,bz, wc0,wc1,bc

    const int tid  = threadIdx.x;
    const int lane = tid & 31;
    const int wv   = tid >> 5;
    const int m    = lane & 15;     // node-in-group (A rows) / col-in-tile (B,C)
    const int hi   = (lane >= 16) ? 1 : 0;

    // ---- Stage 0: stage weights/biases in LDS (block-cooperative) ----------
    for (int i = tid; i < 2 * H2v * H1v; i += BLOCK_THREADS) {
        int sel  = i >> 12;         // 0 = z block, 1 = c block
        int rem  = i & 4095;
        int col  = rem >> 5;        // 0..127
        int k    = rem & 31;        // 0..31
        int scol = sel ? (256 + col) : col;
        sW[i] = (_Float16)Wx2[k * 384 + scol];
    }
    if (tid < H2v) {
        sBz[tid] = bx2[tid]       + bh2[tid];
        sBc[tid] = bx2[256 + tid] + bh2[256 + tid];
        sLw[tid] = lin_w[tid];
    }
    if (tid < H1v) {
        int j = tid, jc = 64 + tid;
        sL1[0][j] = Wx1[j];        sL1[1][j] = Wx1[96 + j];
        sL1[2][j] = bx1[j]  + bh1[j];
        sL1[3][j] = Wx1[jc];       sL1[4][j] = Wx1[96 + jc];
        sL1[5][j] = bx1[jc] + bh1[jc];
    }
    __syncthreads();

    const int g = blockIdx.x * WAVES_PER_BLOCK + wv;
    if (g >= nGroups) return;

    // ---- Stage 1: layer-1 GRU (h=0) in VALU, write a1 tile [16][32] f16 ----
    {
        const int node = g * 16 + m;
        float x0 = 0.0f, x1 = 0.0f;
        if (node < nNodes) {
            float2 xv = ((const float2*)x)[node];
            x0 = xv.x; x1 = xv.y;
        }
        const int fbase = hi * 16;  // lanes 0-15: feats 0..15; 16-31: 16..31
        #pragma unroll
        for (int f0 = 0; f0 < 16; ++f0) {
            int j = fbase + f0;     // hidden feature (0..31)
            float xz  = fmaf(x0, sL1[0][j], fmaf(x1, sL1[1][j], sL1[2][j]));
            float xc  = fmaf(x0, sL1[3][j], fmaf(x1, sL1[4][j], sL1[5][j]));
            float omz = one_minus_sigmoid_f(xz);
            float c   = tanh_f(xc);
            float h1  = omz * c;
            sA[wv][m * H1v + j] = (_Float16)fmaxf(h1, 0.0f);   // relu
        }
    }
    __syncthreads();   // uniform: every live wave hits this exactly once

    // ---- Stage 2: gather A fragment per ISA 16-bit 16x32 A layout ----------
    // pair p lives in vgpr p; K = (p/4)*16 + 8*(lane>=16) + 2*(p%4)
    v16h afrag;
    {
        const _Float16* ap = &sA[wv][m * H1v];
        #pragma unroll
        for (int p = 0; p < 8; ++p) {
            int K = ((p >> 2) << 4) + (hi << 3) + ((p & 3) << 1);
            v2h pr = *(const v2h*)(ap + K);
            afrag[2 * p]     = pr[0];
            afrag[2 * p + 1] = pr[1];
        }
    }

    // head scalars (wave-uniform)
    const float lb0 = lin_b[0];
    const float gw0 = grad_w[0], gw1 = grad_w[1];
    const float gb0 = grad_b[0], gb1 = grad_b[1];

    float acc[8];
    #pragma unroll
    for (int j = 0; j < 8; ++j) acc[j] = 0.0f;

    // ---- Stage 3: 8 tiles x (z-WMMA + c-WMMA) + fused gates + head partials
    #pragma unroll
    for (int t = 0; t < 8; ++t) {
        const int col = t * 16 + m;            // hidden feature 0..127
        // B fragment: element e = W[K = 8*hi + e (e<8), 16 + 8*hi + (e-8) (e>=8)][col]
        const _Float16* bzp = &sW[0 * 4096 + col * H1v + hi * 8];
        const _Float16* bcp = &sW[1 * 4096 + col * H1v + hi * 8];
        v8h bzl = ((const v8h*)bzp)[0], bzh = ((const v8h*)bzp)[2];
        v8h bcl = ((const v8h*)bcp)[0], bch = ((const v8h*)bcp)[2];
        v16h bz, bc;
        #pragma unroll
        for (int i = 0; i < 8; ++i) {
            bz[i] = bzl[i]; bz[i + 8] = bzh[i];
            bc[i] = bcl[i]; bc[i + 8] = bch[i];
        }

        v8f c0 = {};
        v8f dz = __builtin_amdgcn_wmma_f32_16x16x32_f16(
                     false, afrag, false, bz, (short)0, c0, false, false);
        v8f dc = __builtin_amdgcn_wmma_f32_16x16x32_f16(
                     false, afrag, false, bc, (short)0, c0, false, false);

        const float bzs = sBz[col];
        const float bcs = sBc[col];
        const float lw  = sLw[col];

        #pragma unroll
        for (int j = 0; j < 8; ++j) {
            // lane holds D[node = j + 8*hi][feat = col]
            float omz = one_minus_sigmoid_f(dz[j] + bzs);
            float c   = tanh_f(dc[j] + bcs);
            float h2  = omz * c;
            acc[j] = fmaf(tanh_f(h2), lw, acc[j]);
        }
    }

    // ---- Stage 4: reduce the 128 features across each 16-lane half ---------
    #pragma unroll
    for (int mask = 1; mask < 16; mask <<= 1) {
        #pragma unroll
        for (int j = 0; j < 8; ++j)
            acc[j] += __shfl_xor(acc[j], mask, 32);
    }

    // ---- Stage 5: head + J-rotation, writeback -----------------------------
    // out[:,0] = dh[:,1] = Hv*gw1+gb1 ; out[:,1] = -dh[:,0] = -(Hv*gw0+gb0)
    if ((lane & 15) == 0) {
        const int nodeBase = g * 16 + (hi ? 8 : 0);
        #pragma unroll
        for (int j = 0; j < 8; ++j) {
            int node = nodeBase + j;
            if (node < nNodes) {
                float Hv = acc[j] + lb0;
                out[2 * (size_t)node + 0] =  fmaf(Hv, gw1, gb1);
                out[2 * (size_t)node + 1] = -fmaf(Hv, gw0, gb0);
            }
        }
    }
}

extern "C" void kernel_launch(void* const* d_in, const int* in_sizes, int n_in,
                              void* d_out, int out_size, void* d_ws, size_t ws_size,
                              hipStream_t stream) {
    // setup_inputs order:
    // 0:x 1:edge_index 2:Wx1 3:bx1 4:Wh1 5:bh1 6:Wx2 7:bx2 8:Wh2 9:bh2
    // 10:lin_w 11:lin_b 12:grad_w 13:grad_b
    const float* x     = (const float*)d_in[0];
    const float* Wx1   = (const float*)d_in[2];
    const float* bx1   = (const float*)d_in[3];
    const float* bh1   = (const float*)d_in[5];
    const float* Wx2   = (const float*)d_in[6];
    const float* bx2   = (const float*)d_in[7];
    const float* bh2   = (const float*)d_in[9];
    const float* lin_w = (const float*)d_in[10];
    const float* lin_b = (const float*)d_in[11];
    const float* gw    = (const float*)d_in[12];
    const float* gb    = (const float*)d_in[13];
    float* out = (float*)d_out;

    const int nNodes  = in_sizes[0] / 2;           // x is [N, 2]
    const int nGroups = (nNodes + 15) / 16;
    const int blocks  = (nGroups + WAVES_PER_BLOCK - 1) / WAVES_PER_BLOCK;

    hipLaunchKernelGGL(ggcnn_hnn_fused, dim3(blocks), dim3(BLOCK_THREADS), 0, stream,
                       x, Wx1, bx1, bh1, Wx2, bx2, bh2,
                       lin_w, lin_b, gw, gb, out, nNodes, nGroups);
}
